// FCEventuallyLayer_82205674045458
// MI455X (gfx1250) — matrix operand order repaired and checked by reading, not compiled, not run on hardware
//
#include <hip/hip_runtime.h>

// out = relu(1 - beta + X @ W^T),  X,W: [4096,4096] f32, out f32.
// Phase 1: convert X,W -> bf16 in d_ws (each element exactly once, bandwidth-bound).
// Phase 2: bf16 WMMA GEMM (v_wmma_f32_16x16x32_bf16, f32 accum), 128x128 tile,
//          8 waves (2x4), 64x32/wave, BK=64 double-buffered LDS.
//          Staging via GLOBAL_LOAD_ASYNC_TO_LDS_B128 (ASYNCcnt) -> no staging VGPRs,
//          no spills, DMA for stage k+1 overlaps WMMA on stage k.

typedef __attribute__((ext_vector_type(16))) __bf16 v16bf;
typedef __attribute__((ext_vector_type(8)))  float  v8f;
typedef __attribute__((ext_vector_type(4)))  int    i32x4;

#define NDIM   4096
#define BM     128
#define BN     128
#define BK     64
#define LDK    72    // padded bf16 row stride: 144 B (16B aligned, bank-conflict-free)
#define NSTAGE (NDIM / BK)

#if __has_builtin(__builtin_amdgcn_global_load_async_to_lds_b128)
#define USE_ASYNC_LDS 1
#else
#define USE_ASYNC_LDS 0
#endif

#if __has_builtin(__builtin_amdgcn_s_wait_asynccnt)
#define WAIT_ASYNC() __builtin_amdgcn_s_wait_asynccnt(0)
#else
#define WAIT_ASYNC() asm volatile("s_wait_asynccnt 0x0" ::: "memory")
#endif

__device__ __forceinline__ unsigned int pk_bf16(float a, float b) {
#if __has_builtin(__builtin_amdgcn_cvt_pk_bf16_f32)
    typedef __attribute__((ext_vector_type(2))) __bf16 v2bf;
    union { v2bf v; unsigned int u; } r;
    r.v = __builtin_amdgcn_cvt_pk_bf16_f32(a, b);
    return r.u;
#else
    unsigned int ua = __float_as_uint(a);
    ua = (ua + 0x7FFFu + ((ua >> 16) & 1u)) >> 16;
    unsigned int ub = __float_as_uint(b);
    ub = (ub + 0x7FFFu + ((ub >> 16) & 1u)) >> 16;
    return ua | (ub << 16);
#endif
}

// ---------------- Phase 1: f32 -> bf16 conversion (one pass, each elem once) ---
__global__ __launch_bounds__(256)
void cvt_f32_to_bf16_kernel(const float4* __restrict__ src,
                            uint4* __restrict__ dst, int n8) {
    const int i = blockIdx.x * blockDim.x + threadIdx.x;
    if (i >= n8) return;
    const float4 a = src[2 * i];
    const float4 b = src[2 * i + 1];
    uint4 o;
    o.x = pk_bf16(a.x, a.y);
    o.y = pk_bf16(a.z, a.w);
    o.z = pk_bf16(b.x, b.y);
    o.w = pk_bf16(b.z, b.w);
    dst[i] = o;
}

// ---------------- Phase 2: bf16 WMMA GEMM + bias + relu ------------------------
__device__ __forceinline__ v16bf load_frag(const unsigned short* p, int second_off_ushorts) {
    union { v16bf v; uint4 q[2]; } u;
    u.q[0] = *(const uint4*)(p);
    u.q[1] = *(const uint4*)(p + second_off_ushorts);
    return u.v;
}

__global__ __launch_bounds__(256, 1)
void fc_relu_wmma_kernel(const unsigned short* __restrict__ Xb,
                         const unsigned short* __restrict__ Wb,
                         const float* __restrict__ Beta,
                         float* __restrict__ Out) {
    __shared__ __align__(16) unsigned short As[2][BM * LDK];
    __shared__ __align__(16) unsigned short Bs[2][BN * LDK];

    const int tid    = threadIdx.x;
    const int lane   = tid & 31;
    const int wave   = tid >> 5;
    const int lane16 = lane & 15;
    const int half   = lane >> 4;

    const int wm = (wave >> 2) * 64;  // wave M base within block tile (0 or 64)
    const int wn = (wave & 3)  * 32;  // wave N base within block tile

    const int bm = blockIdx.y * BM;   // batch rows
    const int bn = blockIdx.x * BN;   // output cols (rows of W)

    // staging roles: 8 uint4-chunks per 64-elem bf16 row -> 8 threads/row, 32 rows/pass
    const int tr = tid >> 3;          // 0..31
    const int tc = tid & 7;           // 16B chunk within row (8 bf16)

    const v8f vzero = {0.f, 0.f, 0.f, 0.f, 0.f, 0.f, 0.f, 0.f};
    v8f acc[4][2];
#pragma unroll
    for (int i = 0; i < 4; ++i)
#pragma unroll
        for (int j = 0; j < 2; ++j) acc[i][j] = vzero;

#if USE_ASYNC_LDS
    auto prefetch_async = [&](int ks, int buf) {
        const int k0 = ks * BK;
#pragma unroll
        for (int i = 0; i < 4; ++i) {
            const int row = tr + i * 32;
            const unsigned short* ga = Xb + (size_t)(bm + row) * NDIM + k0 + tc * 8;
            const unsigned short* gb = Wb + (size_t)(bn + row) * NDIM + k0 + tc * 8;
            const int loff = row * LDK + tc * 8;
            __builtin_amdgcn_global_load_async_to_lds_b128(
                (i32x4*)ga, (i32x4*)&As[buf][loff], 0, 0);
            __builtin_amdgcn_global_load_async_to_lds_b128(
                (i32x4*)gb, (i32x4*)&Bs[buf][loff], 0, 0);
        }
    };
#else
    auto copy_stage = [&](int ks, int buf) {
        const int k0 = ks * BK;
#pragma unroll
        for (int i = 0; i < 4; ++i) {
            const int row = tr + i * 32;
            uint4 a = *(const uint4*)(Xb + (size_t)(bm + row) * NDIM + k0 + tc * 8);
            uint4 b = *(const uint4*)(Wb + (size_t)(bn + row) * NDIM + k0 + tc * 8);
            const int loff = row * LDK + tc * 8;
            *(uint4*)&As[buf][loff] = a;
            *(uint4*)&Bs[buf][loff] = b;
        }
    };
#endif

    auto compute = [&](int buf) {
#pragma unroll
        for (int kk = 0; kk < BK; kk += 32) {
            v16bf af[4], bq[2];
            // A frag (16x32): elems 0-7 -> K=kk+8h+e ; 8-15 -> K=kk+16+8h+(e-8)
#pragma unroll
            for (int tm = 0; tm < 4; ++tm) {
                const int row = wm + tm * 16 + lane16;
                af[tm] = load_frag(&As[buf][row * LDK + kk + half * 8], 16);
            }
            // B frag (32x16): elems 0-7 -> K=kk+8h+e ; 8-15 -> K=kk+16+8h+(e-8)
#pragma unroll
            for (int tn = 0; tn < 2; ++tn) {
                const int row = wn + tn * 16 + lane16;
                bq[tn] = load_frag(&Bs[buf][row * LDK + kk + half * 8], 16);
            }
#pragma unroll
            for (int tm = 0; tm < 4; ++tm)
#pragma unroll
                for (int tn = 0; tn < 2; ++tn)
                    acc[tm][tn] = __builtin_amdgcn_wmma_f32_16x16x32_bf16(
                        false, af[tm], false, bq[tn],
                        (short)0, acc[tm][tn], false, false);
        }
    };

#if USE_ASYNC_LDS
    prefetch_async(0, 0);
    WAIT_ASYNC();
    __syncthreads();
    for (int ks = 0; ks < NSTAGE; ++ks) {
        const int buf   = ks & 1;
        const bool more = (ks + 1 < NSTAGE);
        if (more) prefetch_async(ks + 1, buf ^ 1);  // async DMA overlaps WMMA
        compute(buf);
        if (more) {
            WAIT_ASYNC();      // own async copies into buf^1 complete
            __syncthreads();   // all waves' copies visible
        }
    }
#else
    copy_stage(0, 0);
    __syncthreads();
    for (int ks = 0; ks < NSTAGE; ++ks) {
        const int buf = ks & 1;
        compute(buf);
        if (ks + 1 < NSTAGE) {
            __syncthreads();
            copy_stage(ks + 1, buf ^ 1);
            __syncthreads();
        }
    }
#endif

    const float bias = 1.0f - Beta[0];
#pragma unroll
    for (int tm = 0; tm < 4; ++tm) {
#pragma unroll
        for (int tn = 0; tn < 2; ++tn) {
            const int col = bn + wn + tn * 16 + lane16;
            const size_t base = (size_t)(bm + wm + tm * 16 + half * 8) * NDIM + col;
#pragma unroll
            for (int r = 0; r < 8; ++r) {
                const float v = acc[tm][tn][r] + bias;
                Out[base + (size_t)r * NDIM] = fmaxf(v, 0.0f);
            }
        }
    }
}

extern "C" void kernel_launch(void* const* d_in, const int* in_sizes, int n_in,
                              void* d_out, int out_size, void* d_ws, size_t ws_size,
                              hipStream_t stream) {
    const float* x    = (const float*)d_in[0];
    const float* w    = (const float*)d_in[1];
    const float* beta = (const float*)d_in[2];
    float* out = (float*)d_out;

    unsigned short* Xb = (unsigned short*)d_ws;                       // 32 MB
    unsigned short* Wb = Xb + (size_t)NDIM * NDIM;                    // 32 MB

    const int n8 = (NDIM * NDIM) / 8;   // 8 floats per thread
    const int cvt_blocks = n8 / 256;
    cvt_f32_to_bf16_kernel<<<cvt_blocks, 256, 0, stream>>>(
        (const float4*)x, (uint4*)Xb, n8);
    cvt_f32_to_bf16_kernel<<<cvt_blocks, 256, 0, stream>>>(
        (const float4*)w, (uint4*)Wb, n8);

    dim3 grid(NDIM / BN, NDIM / BM);
    dim3 block(256);
    fc_relu_wmma_kernel<<<grid, block, 0, stream>>>(Xb, Wb, beta, out);
}
